// SchNet_44401371906217
// MI455X (gfx1250) — matrix operand order repaired and checked by reading, not compiled, not run on hardware
//
#include <hip/hip_runtime.h>

// ---------------- problem constants (match reference) ----------------
#define NNODES 50000
#define NEDGES 1600000
#define HIDD   128
#define NFD    128
#define NGD    50
#define NGP    64      // NG padded to a WMMA K multiple
#define NLD    6
#define NSEGD  64
#define CUT    20.0f
#define EDGE_BLOCKS 2048

typedef __attribute__((ext_vector_type(16))) _Float16 v16h;
typedef __attribute__((ext_vector_type(8)))  float    v8f;

// ---------------- helpers ----------------
__device__ __forceinline__ float dssp(float x) {
    // shifted softplus, numerically stable: softplus(x) - ln 2
    float ax = fabsf(x);
    return fmaxf(x, 0.f) + __logf(1.f + __expf(-ax)) - 0.69314718055994531f;
}

__device__ __forceinline__ v8f wmma16(v16h a, v16h b, v8f c) {
    // v_wmma_f32_16x16x32_f16 : D = A(16x32 f16) * B(32x16 f16) + C(f32)
    return __builtin_amdgcn_wmma_f32_16x16x32_f16(false, a, false, b, (short)0, c, false, false);
}

__device__ __forceinline__ void wait_asynccnt0() {
#if __has_builtin(__builtin_amdgcn_s_wait_asynccnt)
    __builtin_amdgcn_s_wait_asynccnt(0);
#else
    asm volatile("s_wait_asynccnt 0x0" ::: "memory");
#endif
}

// async copy of 16 bytes/lane: global (saddr + voffset) -> LDS (byte addr in VGPR)
__device__ __forceinline__ void async_copy_b128(unsigned ldsAddr, unsigned gOff,
                                                const void* sbase) {
    asm volatile("global_load_async_to_lds_b128 %0, %1, %2"
                 :: "v"(ldsAddr), "v"(gOff), "s"(sbase)
                 : "memory");
}

union Frag32B { int4 i4[2]; v16h v; };

// A stored row-major [M][K] (K contiguous), LDS or global.
// lane L: M = L%16, h = L/16; halves i: K = (i/8)*16 + h*8 + i%8
__device__ __forceinline__ v16h loadA(const _Float16* base, int lda, int m0, int k0, int lane) {
    Frag32B f;
    const _Float16* p = base + (m0 + (lane & 15)) * lda + k0 + ((lane >> 4) << 3);
    f.i4[0] = *(const int4*)(p);        // K = k0 + h*8 + [0..7]
    f.i4[1] = *(const int4*)(p + 16);   // K = k0 + 16 + h*8 + [0..7]
    return f.v;
}

// B stored transposed [N][K] (K contiguous), LDS or global.
// lane L: N = L%16, h = L/16; halves i: K = (i/8)*16 + h*8 + i%8
__device__ __forceinline__ v16h loadB(const _Float16* base, int ldb, int n0, int k0, int lane) {
    Frag32B f;
    const _Float16* p = base + (n0 + (lane & 15)) * ldb + k0 + ((lane >> 4) << 3);
    f.i4[0] = *(const int4*)(p);
    f.i4[1] = *(const int4*)(p + 16);
    return f.v;
}

// ---------------- prep kernels ----------------
// W [K,N] f32 row-major  ->  Wt [N,Kpad] f16 (zero-padded K)
__global__ void k_transpose_f16(const float* __restrict__ W, _Float16* __restrict__ Wt,
                                int K, int N, int Kpad) {
    int i = blockIdx.x * 256 + threadIdx.x;
    if (i >= N * Kpad) return;
    int n = i / Kpad, k = i - n * Kpad;
    Wt[i] = (k < K) ? (_Float16)W[k * N + n] : (_Float16)0.f;
}

__global__ void k_zero(float* __restrict__ p, long long n) {
    long long i = (long long)blockIdx.x * 256 + threadIdx.x;
    if (i < n) p[i] = 0.f;
}

// per-edge: distance, cosine cutoff, Gaussian smearing -> fp16 [E][64]
__global__ __launch_bounds__(256) void k_edge_geom(
    const float* __restrict__ pos, const long long* __restrict__ srcI,
    const long long* __restrict__ dstI, _Float16* __restrict__ ea16,
    float* __restrict__ Cf, int nE) {
    long long e = (long long)blockIdx.x * 256 + threadIdx.x;
    if (e >= nE) return;
    int s = (int)srcI[e], t = (int)dstI[e];
    float dx = pos[s * 3 + 0] - pos[t * 3 + 0];
    float dy = pos[s * 3 + 1] - pos[t * 3 + 1];
    float dz = pos[s * 3 + 2] - pos[t * 3 + 2];
    float d = sqrtf(dx * dx + dy * dy + dz * dz);
    Cf[e] = 0.5f * (__cosf(d * (3.14159265358979f / CUT)) + 1.f);
    const float step  = CUT / (float)(NGD - 1);
    const float coeff = -0.5f / (step * step);
    __align__(16) _Float16 buf[NGP];
    #pragma unroll
    for (int g = 0; g < NGP; ++g) {
        float dv = d - step * (float)g;
        buf[g] = (g < NGD) ? (_Float16)__expf(coeff * dv * dv) : (_Float16)0.f;
    }
    int4* d4 = (int4*)(ea16 + (size_t)e * NGP);
    const int4* s4 = (const int4*)buf;
    #pragma unroll
    for (int i = 0; i < 8; ++i) d4[i] = s4[i];
}

// ---------------- generic node GEMM: Y[opt +=] act(X @ W + b) ----------------
// 256 threads = 8 waves; block = 32 rows x 128 cols; wave w -> N-tile w.
// B fragments loaded straight from global (L2-hot, 32 KB weights) into registers.
__global__ __launch_bounds__(256) void k_node_gemm(
    const float* __restrict__ X, const _Float16* __restrict__ Wt,
    const float* __restrict__ bias, float* __restrict__ Y,
    int nRows, int applySSP, int residual) {
    __shared__ __align__(16) _Float16 Xs[32 * 128];
    const int tid = threadIdx.x, lane = tid & 31, wv = tid >> 5;
    const int row0 = blockIdx.x * 32;
    const int n0 = wv << 4;

    v16h bw[4];
    #pragma unroll
    for (int kk = 0; kk < 4; ++kk) bw[kk] = loadB(Wt, 128, n0, kk * 32, lane);

    for (int i = tid; i < 32 * 128; i += 256) { // stage X tile, f32->f16
        int row = row0 + (i >> 7);
        Xs[i] = (row < nRows) ? (_Float16)X[(size_t)row * 128 + (i & 127)] : (_Float16)0.f;
    }
    __syncthreads();

    v8f acc0 = {}, acc1 = {};
    #pragma unroll
    for (int kk = 0; kk < 4; ++kk) {
        v16h a0 = loadA(Xs, 128,  0, kk * 32, lane);
        v16h a1 = loadA(Xs, 128, 16, kk * 32, lane);
        acc0 = wmma16(a0, bw[kk], acc0);
        acc1 = wmma16(a1, bw[kk], acc1);
    }

    const int n  = n0 + (lane & 15);
    const int mh = (lane >> 4) << 3;
    const float bn = bias ? bias[n] : 0.f;
    #pragma unroll
    for (int v = 0; v < 8; ++v) {
        int r0 = row0 + mh + v;
        if (r0 < nRows) {
            float o = acc0[v] + bn; if (applySSP) o = dssp(o);
            size_t idx = (size_t)r0 * 128 + n;
            if (residual) Y[idx] += o; else Y[idx] = o;
        }
        int r1 = row0 + 16 + mh + v;
        if (r1 < nRows) {
            float o = acc1[v] + bn; if (applySSP) o = dssp(o);
            size_t idx = (size_t)r1 * 128 + n;
            if (residual) Y[idx] += o; else Y[idx] = o;
        }
    }
}

// ---------------- fused edge kernel (persistent blocks) ----------------
// Each block loops over 32-edge tiles. Filter weights live in REGISTERS for the
// whole loop (loop-invariant B fragments). Edge-attr tiles are double-buffered
// in LDS and prefetched with global_load_async_to_lds_b128 (ASYNCcnt).
__global__ __launch_bounds__(256) void k_edge_cfconv(
    const _Float16* __restrict__ ea16, const float* __restrict__ Cf,
    const long long* __restrict__ srcI, const long long* __restrict__ dstI,
    const _Float16* __restrict__ W1t, const float* __restrict__ b1,
    const _Float16* __restrict__ W2t, const float* __restrict__ b2,
    const float* __restrict__ hx, float* __restrict__ agg, int ntiles) {
    __shared__ __align__(16) _Float16 Eas[2][32 * 64];   // 2 x 4 KB
    __shared__ __align__(16) _Float16 Ts[32 * 128];      // 8 KB
    __shared__ int   esrc[2][32];
    __shared__ int   edst[2][32];
    __shared__ float eC[2][32];

    const int tid = threadIdx.x, lane = tid & 31, wv = tid >> 5;
    const int n0 = wv << 4;
    const int n  = n0 + (lane & 15);
    const int mh = (lane >> 4) << 3;

    // register-resident weight fragments (stationary across all tiles)
    v16h bw1[2], bw2[4];
    #pragma unroll
    for (int kk = 0; kk < 2; ++kk) bw1[kk] = loadB(W1t, 64, n0, kk * 32, lane);
    #pragma unroll
    for (int kk = 0; kk < 4; ++kk) bw2[kk] = loadB(W2t, 128, n0, kk * 32, lane);
    const float b1n = b1[n], b2n = b2[n];

    const unsigned eaLds = (unsigned)(uintptr_t)&Eas[0][0];
    // prologue: async-prefetch first tile into buffer 0 (each lane moves 16 B)
    async_copy_b128(eaLds + (unsigned)tid * 16u,
                    (unsigned)blockIdx.x * 4096u + (unsigned)tid * 16u, ea16);

    int j = 0;
    for (int t = blockIdx.x; t < ntiles; t += gridDim.x, ++j) {
        const int p = j & 1;
        // per-edge scalars for current tile (double-buffered)
        if (tid < 32) {
            long long e = (long long)t * 32 + tid;
            esrc[p][tid] = (int)srcI[e];
            edst[p][tid] = (int)dstI[e];
            eC[p][tid]   = Cf[e];
        }
        // Ea[p] landed?  (one outstanding async per wave)
        wait_asynccnt0();
        __syncthreads();
        // all waves past the barrier -> safe to overwrite Ea[p^1]; prefetch next tile
        {
            int tn = t + gridDim.x; if (tn >= ntiles) tn = t;   // harmless tail reload
            async_copy_b128(eaLds + (unsigned)((p ^ 1) * (32 * 64 * 2)) + (unsigned)tid * 16u,
                            (unsigned)tn * 4096u + (unsigned)tid * 16u, ea16);
        }

        // GEMM1: [32,64] @ [64,128]
        v8f u0 = {}, u1 = {};
        #pragma unroll
        for (int kk = 0; kk < 2; ++kk) {
            v16h a0 = loadA(&Eas[p][0], 64,  0, kk * 32, lane);
            v16h a1 = loadA(&Eas[p][0], 64, 16, kk * 32, lane);
            u0 = wmma16(a0, bw1[kk], u0);
            u1 = wmma16(a1, bw1[kk], u1);
        }
        #pragma unroll
        for (int v = 0; v < 8; ++v) {
            Ts[(mh + v) * 128 + n]      = (_Float16)dssp(u0[v] + b1n);
            Ts[(16 + mh + v) * 128 + n] = (_Float16)dssp(u1[v] + b1n);
        }
        __syncthreads();

        // GEMM2: [32,128] @ [128,128]
        v8f w0 = {}, w1 = {};
        #pragma unroll
        for (int kk = 0; kk < 4; ++kk) {
            v16h a0 = loadA(Ts, 128,  0, kk * 32, lane);
            v16h a1 = loadA(Ts, 128, 16, kk * 32, lane);
            w0 = wmma16(a0, bw2[kk], w0);
            w1 = wmma16(a1, bw2[kk], w1);
        }
        // epilogue: msg = W * C * hx[src] (gather, L2-resident), scatter-add
        #pragma unroll
        for (int v = 0; v < 8; ++v) {
            {
                int em = mh + v;
                float Wv = (w0[v] + b2n) * eC[p][em];
                float m  = Wv * hx[(size_t)esrc[p][em] * 128 + n];
                atomicAdd(&agg[(size_t)edst[p][em] * 128 + n], m);
            }
            {
                int em = 16 + mh + v;
                float Wv = (w1[v] + b2n) * eC[p][em];
                float m  = Wv * hx[(size_t)esrc[p][em] * 128 + n];
                atomicAdd(&agg[(size_t)edst[p][em] * 128 + n], m);
            }
        }
        // next iteration's leading barrier separates Ts/Ea reuse
    }
}

// ---------------- readout: out[batch] += ssp(h@out1+b1)@out2 + b2 ----------------
__global__ __launch_bounds__(256) void k_readout(
    const float* __restrict__ h, const _Float16* __restrict__ O1t,
    const float* __restrict__ b1, const float* __restrict__ w2,
    const float* __restrict__ b2, const long long* __restrict__ batch,
    float* __restrict__ out, int nRows) {
    __shared__ __align__(16) _Float16 Xs[32 * 128];   // 8 KB
    __shared__ float red[32 * 64];                    // 8 KB
    const int tid = threadIdx.x, lane = tid & 31, wv = tid >> 5;
    const int row0 = blockIdx.x * 32;
    const int mt = wv >> 2;            // 8 waves = 2 M-tiles x 4 N-tiles (64 cols)
    const int n0 = (wv & 3) << 4;

    v16h bw[4];
    #pragma unroll
    for (int kk = 0; kk < 4; ++kk) bw[kk] = loadB(O1t, 128, n0, kk * 32, lane);

    for (int i = tid; i < 32 * 128; i += 256) {
        int row = row0 + (i >> 7);
        Xs[i] = (row < nRows) ? (_Float16)h[(size_t)row * 128 + (i & 127)] : (_Float16)0.f;
    }
    __syncthreads();

    v8f acc = {};
    #pragma unroll
    for (int kk = 0; kk < 4; ++kk) {
        v16h a = loadA(Xs, 128, mt * 16, kk * 32, lane);
        acc = wmma16(a, bw[kk], acc);
    }
    const int n  = n0 + (lane & 15);
    const int mh = (lane >> 4) << 3;
    const float b1n = b1[n], w2n = w2[n];
    #pragma unroll
    for (int v = 0; v < 8; ++v) {
        int em = mt * 16 + mh + v;
        red[em * 64 + n] = dssp(acc[v] + b1n) * w2n;
    }
    __syncthreads();
    if (tid < 32) {
        int row = row0 + tid;
        if (row < nRows) {
            float s = b2[0];
            #pragma unroll
            for (int jj = 0; jj < 64; ++jj) s += red[tid * 64 + jj];
            atomicAdd(&out[(int)batch[row]], s);
        }
    }
}

// ---------------- launcher ----------------
extern "C" void kernel_launch(void* const* d_in, const int* in_sizes, int n_in,
                              void* d_out, int out_size, void* d_ws, size_t ws_size,
                              hipStream_t stream) {
    const float*     h      = (const float*)d_in[0];
    const float*     pos    = (const float*)d_in[1];
    const long long* eidx   = (const long long*)d_in[2];   // int64 in reference
    const long long* batch  = (const long long*)d_in[3];
    const float*     mlp_w1 = (const float*)d_in[4];
    const float*     mlp_b1 = (const float*)d_in[5];
    const float*     mlp_w2 = (const float*)d_in[6];
    const float*     mlp_b2 = (const float*)d_in[7];
    const float*     cl1_w  = (const float*)d_in[8];
    const float*     cl2_w  = (const float*)d_in[9];
    const float*     cl2_b  = (const float*)d_in[10];
    const float*     lin_w  = (const float*)d_in[11];
    const float*     lin_b  = (const float*)d_in[12];
    const float*     out1_w = (const float*)d_in[13];
    const float*     out1_b = (const float*)d_in[14];
    const float*     out2_w = (const float*)d_in[15];
    const float*     out2_b = (const float*)d_in[16];
    const long long* srcI = eidx;
    const long long* dstI = eidx + NEDGES;

    // workspace carve-out (~290 MB)
    char* ws = (char*)d_ws;
    size_t off = 0;
    auto wsalloc = [&](size_t bytes) -> void* {
        void* p = (void*)(ws + off);
        off += (bytes + 255) & ~(size_t)255;
        return p;
    };
    _Float16* ea16 = (_Float16*)wsalloc((size_t)NEDGES * NGP * sizeof(_Float16));
    float*    Cf   = (float*)   wsalloc((size_t)NEDGES * sizeof(float));
    float*    hcur = (float*)   wsalloc((size_t)NNODES * HIDD * sizeof(float));
    float*    hx   = (float*)   wsalloc((size_t)NNODES * HIDD * sizeof(float));
    float*    agg  = (float*)   wsalloc((size_t)NNODES * HIDD * sizeof(float));
    const size_t LW = 64 * 128 + 4 * 128 * 128;   // fp16 elems per layer
    _Float16* w16  = (_Float16*)wsalloc((NLD * LW + 64 * 128) * sizeof(_Float16));

    // 1) weights -> fp16 transposed [N][Kpad]
    for (int l = 0; l < NLD; ++l) {
        _Float16* base = w16 + (size_t)l * LW;
        k_transpose_f16<<<(128 * 64 + 255) / 256, 256, 0, stream>>>(
            mlp_w1 + (size_t)l * NGD * NFD, base, NGD, NFD, 64);
        k_transpose_f16<<<(128 * 128 + 255) / 256, 256, 0, stream>>>(
            mlp_w2 + (size_t)l * NFD * NFD, base + 8192, NFD, NFD, 128);
        k_transpose_f16<<<(128 * 128 + 255) / 256, 256, 0, stream>>>(
            cl1_w + (size_t)l * HIDD * NFD, base + 8192 + 16384, HIDD, NFD, 128);
        k_transpose_f16<<<(128 * 128 + 255) / 256, 256, 0, stream>>>(
            cl2_w + (size_t)l * NFD * HIDD, base + 8192 + 32768, NFD, HIDD, 128);
        k_transpose_f16<<<(128 * 128 + 255) / 256, 256, 0, stream>>>(
            lin_w + (size_t)l * HIDD * HIDD, base + 8192 + 49152, HIDD, HIDD, 128);
    }
    k_transpose_f16<<<(64 * 128 + 255) / 256, 256, 0, stream>>>(
        out1_w, w16 + (size_t)NLD * LW, HIDD, 64, 128);

    // 2) init h_cur, edge geometry + Gaussian smearing (reused across all 6 layers)
    hipMemcpyAsync(hcur, h, (size_t)NNODES * HIDD * sizeof(float),
                   hipMemcpyDeviceToDevice, stream);
    k_edge_geom<<<(NEDGES + 255) / 256, 256, 0, stream>>>(pos, srcI, dstI, ea16, Cf, NEDGES);

    // 3) interaction layers
    const int nodeBlocks = (NNODES + 31) / 32;
    const int ntiles = NEDGES / 32;
    for (int l = 0; l < NLD; ++l) {
        _Float16* base = w16 + (size_t)l * LW;
        // hx = h @ lin1 (no bias)
        k_node_gemm<<<nodeBlocks, 256, 0, stream>>>(hcur, base + 8192 + 16384,
                                                    nullptr, hx, NNODES, 0, 0);
        k_zero<<<(int)(((long long)NNODES * HIDD + 255) / 256), 256, 0, stream>>>(
            agg, (long long)NNODES * HIDD);
        // W = ssp(ea@W1+b1)@W2+b2; msg = W*C*hx[src]; agg[dst] += msg
        k_edge_cfconv<<<EDGE_BLOCKS, 256, 0, stream>>>(
            ea16, Cf, srcI, dstI,
            base, mlp_b1 + (size_t)l * NFD,
            base + 8192, mlp_b2 + (size_t)l * NFD, hx, agg, ntiles);
        // tmp = ssp(agg @ lin2 + b)   (reuse hx buffer)
        k_node_gemm<<<nodeBlocks, 256, 0, stream>>>(agg, base + 8192 + 32768,
                                                    cl2_b + (size_t)l * HIDD, hx, NNODES, 1, 0);
        // h += tmp @ lin + b
        k_node_gemm<<<nodeBlocks, 256, 0, stream>>>(hx, base + 8192 + 49152,
                                                    lin_b + (size_t)l * HIDD, hcur, NNODES, 0, 1);
    }

    // 4) readout
    k_zero<<<1, 256, 0, stream>>>((float*)d_out, NSEGD);
    k_readout<<<nodeBlocks, 256, 0, stream>>>(hcur, w16 + (size_t)NLD * LW,
                                              out1_b, out2_w, out2_b, batch,
                                              (float*)d_out, NNODES);
}
